// MultiheadSelfAttention_48996986913235
// MI455X (gfx1250) — matrix operand order repaired and checked by reading, not compiled, not run on hardware
//
#include <hip/hip_runtime.h>
#include <hip/hip_bf16.h>

#define D_MODEL 1024
#define SEQ     2048
#define NHEAD   16
#define DHEAD   64
#define BATCH   2
#define BS      (BATCH*SEQ)   // 4096 total rows

typedef __bf16 bf16_t;
typedef bf16_t v16bf __attribute__((ext_vector_type(16)));
typedef bf16_t v8bf  __attribute__((ext_vector_type(8)));
typedef float  v8f   __attribute__((ext_vector_type(8)));

// ---------- helpers ----------

static __device__ __forceinline__ unsigned short f2bf(float f) {
  unsigned int u = __float_as_uint(f);
  u += 0x7FFFu + ((u >> 16) & 1u);        // round-to-nearest-even
  return (unsigned short)(u >> 16);
}

static __device__ __forceinline__ v8bf ld8(const unsigned short* p) {
  return *reinterpret_cast<const v8bf*>(p);
}

static __device__ __forceinline__ v8f wmma_bf16(v16bf a, v16bf b, v8f c) {
  // D = A(16x32) * B(32x16) + C, fp32 accumulate
  return __builtin_amdgcn_wmma_f32_16x16x32_bf16(
      /*neg_a=*/false, a, /*neg_b=*/false, b,
      /*c_mod=*/(short)0, c, /*reuse_a=*/false, /*reuse_b=*/false);
}

// Async copy 16B global -> LDS (GLOBAL_LOAD_ASYNC_TO_LDS_B128, ASYNCcnt).
// Low 32 bits of a generic LDS pointer are the wave-relative LDS byte address
// (ISA 10.2 aperture mapping), which is exactly what VDST expects.
static __device__ __forceinline__ void async_ld_b128(const unsigned short* g,
                                                     const unsigned short* l) {
  unsigned lofs = (unsigned)(unsigned long long)l;
  asm volatile("global_load_async_to_lds_b128 %0, %1, off"
               :: "v"(lofs), "v"(g) : "memory");
}

#define WAIT_ASYNCCNT(imm) asm volatile("s_wait_asynccnt " #imm ::: "memory")

// A fragment: 16x32 bf16 tile at `base` (row-major, leading dim ld).
// ISA layout: lane<16 -> K 0..7 (v0..3), 16..23 (v4..7); lane>=16 -> K 8..15, 24..31.
static __device__ __forceinline__ v16bf load_a(const unsigned short* base, int ld, int lane) {
  const int m  = lane & 15;
  const int kl = (lane & 16) ? 8 : 0;
  const unsigned short* r = base + m * ld;
  union { v16bf v; v8bf h[2]; } u;
  u.h[0] = ld8(r + kl);
  u.h[1] = ld8(r + kl + 16);
  return u.v;
}

// B fragment (K x N = 32 x 16) with B[k][n] = src[n*ld + k]
// (src holds N rows of contiguous K values -> W row-major for D = A*W^T).
// ISA layout: lane<16 -> K 0..7, 16..23 (col n=lane); lane>=16 -> K 8..15, 24..31.
static __device__ __forceinline__ v16bf load_bT(const unsigned short* base, int ld, int lane) {
  const int n  = lane & 15;
  const int kl = (lane & 16) ? 8 : 0;
  const unsigned short* r = base + n * ld + kl;
  union { v16bf v; v8bf h[2]; } u;
  u.h[0] = ld8(r);
  u.h[1] = ld8(r + 16);
  return u.v;
}

// B fragment from an LDS-staged tile [64][LDS_PAD] (row-major in e, contiguous k).
#define LDS_PAD 40
static __device__ __forceinline__ v16bf load_b_lds(const unsigned short* tile,
                                                   int t, int lane) {
  const int n  = lane & 15;
  const int kl = (lane & 16) ? 8 : 0;
  const unsigned short* r = tile + (t * 16 + n) * LDS_PAD + kl;
  union { v16bf v; v8bf h[2]; } u;
  u.h[0] = *reinterpret_cast<const v8bf*>(r);
  u.h[1] = *reinterpret_cast<const v8bf*>(r + 16);
  return u.v;
}

// ---------- kernels ----------

__global__ void cvt_kernel(const float* __restrict__ src,
                           unsigned short* __restrict__ dst, int n) {
  int i = (blockIdx.x * blockDim.x + threadIdx.x) * 4;
  if (i + 3 < n) {
    float4 f = *reinterpret_cast<const float4*>(src + i);
    ushort4 o;
    o.x = f2bf(f.x); o.y = f2bf(f.y); o.z = f2bf(f.z); o.w = f2bf(f.w);
    *reinterpret_cast<ushort4*>(dst + i) = o;
  }
}

// Q/K/V projection: D = x * W^T. Block: 128 rows x 64 cols, 8 waves.
// Weight tile (64x32 bf16) staged in LDS via async copy, double-buffered.
// z==0: Q + RoPE; z==1: K + RoPE; z==2: V stored transposed [b,h,dh,s].
__global__ void qkv_proj_kernel(const unsigned short* __restrict__ xb,
                                const unsigned short* __restrict__ wqb,
                                const unsigned short* __restrict__ wkb,
                                const unsigned short* __restrict__ wvb,
                                unsigned short* __restrict__ qbf,
                                unsigned short* __restrict__ kbf,
                                unsigned short* __restrict__ vtb) {
  __shared__ __align__(16) unsigned short btile[2][64][LDS_PAD];

  const int tid   = threadIdx.x;
  const int lane  = tid & 31;
  const int wave  = tid >> 5;
  const int mBase = blockIdx.x * 128 + wave * 16;
  const int eBase = blockIdx.y * 64;
  const int z     = blockIdx.z;
  const unsigned short* w = (z == 0) ? wqb : ((z == 1) ? wkb : wvb);

  // cooperative tile copy: 256 threads x 16B = 64 rows x 32 k
  const int ce = tid >> 2;            // 0..63  (e within tile)
  const int ck = (tid & 3) * 8;       // 0,8,16,24 (k within tile)
  const unsigned short* gsrc = w + (eBase + ce) * D_MODEL + ck;

  async_ld_b128(gsrc, &btile[0][ce][ck]);   // prefetch k-step 0

  v8f acc[4] = {};
  int cur = 0;
  for (int k0 = 0; k0 < D_MODEL; k0 += 32) {
    const bool has_next = (k0 + 32) < D_MODEL;
    if (has_next) {
      async_ld_b128(gsrc + k0 + 32, &btile[cur ^ 1][ce][ck]);
      WAIT_ASYNCCNT(1);      // in-order: current tile's copy complete
    } else {
      WAIT_ASYNCCNT(0);
    }
    __syncthreads();         // all waves' copies visible

    // batch all fragment loads, then issue WMMAs back-to-back
    v16bf a  = load_a(xb + mBase * D_MODEL + k0, D_MODEL, lane);
    v16bf b0 = load_b_lds(&btile[cur][0][0], 0, lane);
    v16bf b1 = load_b_lds(&btile[cur][0][0], 1, lane);
    v16bf b2 = load_b_lds(&btile[cur][0][0], 2, lane);
    v16bf b3 = load_b_lds(&btile[cur][0][0], 3, lane);
    acc[0] = wmma_bf16(a, b0, acc[0]);
    acc[1] = wmma_bf16(a, b1, acc[1]);
    acc[2] = wmma_bf16(a, b2, acc[2]);
    acc[3] = wmma_bf16(a, b3, acc[3]);

    __syncthreads();         // protect buffer overwritten next iteration
    cur ^= 1;
  }

  const int hi = (lane >> 4), n16 = lane & 15;
  if (z < 2) {  // Q or K with fused RoPE
    unsigned short* dst = (z == 0) ? qbf : kbf;
#pragma unroll
    for (int t = 0; t < 4; ++t) {
      const int e  = eBase + t * 16 + n16;
      const int dh = e & (DHEAD - 1);
      const float invf = __expf(-((float)(dh & ~1) * (1.0f / DHEAD)) * 9.210340371976184f);
#pragma unroll
      for (int v = 0; v < 8; ++v) {
        const int m = mBase + v + 8 * hi;
        const int s = m & (SEQ - 1);
        float c, sn;
        __sincosf((float)s * invf, &sn, &c);
        const float val     = acc[t][v];
        const float partner = __shfl_xor(val, 1);
        const float rot = (dh & 1) ? (partner * sn + val * c)
                                   : (val * c - partner * sn);
        dst[m * D_MODEL + e] = f2bf(rot);
      }
    }
  } else {      // V, transposed: vt[(b*D_MODEL + e)*SEQ + s]
#pragma unroll
    for (int t = 0; t < 4; ++t) {
      const int e = eBase + t * 16 + n16;
#pragma unroll
      for (int v = 0; v < 8; ++v) {
        const int m = mBase + v + 8 * hi;
        const int b = m >> 11;
        const int s = m & (SEQ - 1);
        vtb[((size_t)(b * D_MODEL + e)) * SEQ + s] = f2bf(acc[t][v]);
      }
    }
  }
}

// Flash-style causal attention. Block = (b,h, 128 queries): 8 waves x 16 rows.
__global__ void attn_kernel(const unsigned short* __restrict__ qbf,
                            const unsigned short* __restrict__ kbf,
                            const unsigned short* __restrict__ vtb,
                            unsigned short* __restrict__ abf) {
  __shared__ __align__(16) unsigned short pstage[8][16][32];  // per-wave P staging

  const int lane = threadIdx.x & 31;
  const int wave = threadIdx.x >> 5;
  const int h    = blockIdx.y;
  const int b    = blockIdx.z;
  const int q0   = blockIdx.x * 128 + wave * 16;
  const int mrow = b * SEQ + q0;
  const int hi = (lane >> 4), n16 = lane & 15;

  const v16bf qa0 = load_a(qbf + mrow * D_MODEL + h * DHEAD + 0,  D_MODEL, lane);
  const v16bf qa1 = load_a(qbf + mrow * D_MODEL + h * DHEAD + 32, D_MODEL, lane);

  float rowM[8], rowL[8];
#pragma unroll
  for (int v = 0; v < 8; ++v) { rowM[v] = -INFINITY; rowL[v] = 0.0f; }
  v8f o[4] = {};

  const unsigned short* kbase = kbf + (size_t)b * SEQ * D_MODEL + h * DHEAD;   // [s][dh]
  const unsigned short* vbase = vtb + ((size_t)b * D_MODEL + h * DHEAD) * SEQ; // [dh][s]

  const int nblk = ((q0 + 15) >> 5) + 1;
  for (int kb = 0; kb < nblk; ++kb) {
    const int key0 = kb * 32;

    // prefetch next key block (K rows + V columns) while this one computes
    if (kb + 1 < nblk) {
      const int nk = key0 + 32;
      __builtin_prefetch(kbase + (nk + lane) * D_MODEL, 0, 3);
      __builtin_prefetch(vbase + lane * SEQ + nk, 0, 3);
      __builtin_prefetch(vbase + (lane + 32) * SEQ + nk, 0, 3);
    }

    // scores S = Q * K^T: batch all four K fragments, then 4 WMMAs
    v16bf kb00 = load_bT(kbase + (key0 +  0) * D_MODEL + 0,  D_MODEL, lane);
    v16bf kb01 = load_bT(kbase + (key0 +  0) * D_MODEL + 32, D_MODEL, lane);
    v16bf kb10 = load_bT(kbase + (key0 + 16) * D_MODEL + 0,  D_MODEL, lane);
    v16bf kb11 = load_bT(kbase + (key0 + 16) * D_MODEL + 32, D_MODEL, lane);
    v8f sfr[2];
    {
      v8f s8 = {};
      s8 = wmma_bf16(qa0, kb00, s8);
      sfr[0] = wmma_bf16(qa1, kb01, s8);
      v8f s9 = {};
      s9 = wmma_bf16(qa0, kb10, s9);
      sfr[1] = wmma_bf16(qa1, kb11, s9);
    }

    // scale + causal mask
    float pr[2][8];
#pragma unroll
    for (int t = 0; t < 2; ++t)
#pragma unroll
      for (int v = 0; v < 8; ++v) {
        const int qm  = q0 + v + 8 * hi;
        const int key = key0 + t * 16 + n16;
        const float sv = sfr[t][v] * 0.125f;   // 1/sqrt(64)
        pr[t][v] = (key <= qm) ? sv : -INFINITY;
      }

    // online softmax (row stats replicated across each 16-lane half)
#pragma unroll
    for (int v = 0; v < 8; ++v) {
      float mx = fmaxf(pr[0][v], pr[1][v]);
#pragma unroll
      for (int off = 8; off >= 1; off >>= 1) mx = fmaxf(mx, __shfl_xor(mx, off));
      const float mnew = fmaxf(rowM[v], mx);
      const float sc   = __expf(rowM[v] - mnew);
      rowM[v] = mnew;
      const float p0 = __expf(pr[0][v] - mnew);
      const float p1 = __expf(pr[1][v] - mnew);
      pr[0][v] = p0; pr[1][v] = p1;
      float rs = p0 + p1;
#pragma unroll
      for (int off = 8; off >= 1; off >>= 1) rs += __shfl_xor(rs, off);
      rowL[v] = rowL[v] * sc + rs;
#pragma unroll
      for (int t = 0; t < 4; ++t) o[t][v] = o[t][v] * sc;
    }

    // C-layout -> A-layout conversion of P via per-wave LDS (same-wave RAW, in order)
#pragma unroll
    for (int t = 0; t < 2; ++t)
#pragma unroll
      for (int v = 0; v < 8; ++v)
        pstage[wave][v + 8 * hi][t * 16 + n16] = f2bf(pr[t][v]);

    v16bf pa;
    {
      const int m  = lane & 15;
      const int kl = (lane & 16) ? 8 : 0;
      const unsigned short* r = &pstage[wave][m][0];
      union { v16bf v; v8bf h[2]; } u;
      u.h[0] = *reinterpret_cast<const v8bf*>(r + kl);
      u.h[1] = *reinterpret_cast<const v8bf*>(r + kl + 16);
      pa = u.v;
    }

    // O += P * V: batch all four V fragments, then 4 WMMAs
    v16bf vb0 = load_bT(vbase +  0 * SEQ + key0, SEQ, lane);
    v16bf vb1 = load_bT(vbase + 16 * SEQ + key0, SEQ, lane);
    v16bf vb2 = load_bT(vbase + 32 * SEQ + key0, SEQ, lane);
    v16bf vb3 = load_bT(vbase + 48 * SEQ + key0, SEQ, lane);
    o[0] = wmma_bf16(pa, vb0, o[0]);
    o[1] = wmma_bf16(pa, vb1, o[1]);
    o[2] = wmma_bf16(pa, vb2, o[2]);
    o[3] = wmma_bf16(pa, vb3, o[3]);
  }

  // normalize and store attention output (bf16, row-major [m][e])
#pragma unroll
  for (int t = 0; t < 4; ++t)
#pragma unroll
    for (int v = 0; v < 8; ++v) {
      const int m = mrow + v + 8 * hi;
      abf[m * D_MODEL + h * DHEAD + t * 16 + n16] = f2bf(o[t][v] / rowL[v]);
    }
}

// Output projection: out = attn_out * Wo^T, fp32 stores. Same LDS-staged GEMM.
__global__ void out_proj_kernel(const unsigned short* __restrict__ abf,
                                const unsigned short* __restrict__ wob,
                                float* __restrict__ out) {
  __shared__ __align__(16) unsigned short btile[2][64][LDS_PAD];

  const int tid   = threadIdx.x;
  const int lane  = tid & 31;
  const int wave  = tid >> 5;
  const int mBase = blockIdx.x * 128 + wave * 16;
  const int eBase = blockIdx.y * 64;

  const int ce = tid >> 2;
  const int ck = (tid & 3) * 8;
  const unsigned short* gsrc = wob + (eBase + ce) * D_MODEL + ck;

  async_ld_b128(gsrc, &btile[0][ce][ck]);

  v8f acc[4] = {};
  int cur = 0;
  for (int k0 = 0; k0 < D_MODEL; k0 += 32) {
    const bool has_next = (k0 + 32) < D_MODEL;
    if (has_next) {
      async_ld_b128(gsrc + k0 + 32, &btile[cur ^ 1][ce][ck]);
      WAIT_ASYNCCNT(1);
    } else {
      WAIT_ASYNCCNT(0);
    }
    __syncthreads();

    v16bf a  = load_a(abf + mBase * D_MODEL + k0, D_MODEL, lane);
    v16bf b0 = load_b_lds(&btile[cur][0][0], 0, lane);
    v16bf b1 = load_b_lds(&btile[cur][0][0], 1, lane);
    v16bf b2 = load_b_lds(&btile[cur][0][0], 2, lane);
    v16bf b3 = load_b_lds(&btile[cur][0][0], 3, lane);
    acc[0] = wmma_bf16(a, b0, acc[0]);
    acc[1] = wmma_bf16(a, b1, acc[1]);
    acc[2] = wmma_bf16(a, b2, acc[2]);
    acc[3] = wmma_bf16(a, b3, acc[3]);

    __syncthreads();
    cur ^= 1;
  }

  const int hi = (lane >> 4), n16 = lane & 15;
#pragma unroll
  for (int t = 0; t < 4; ++t)
#pragma unroll
    for (int v = 0; v < 8; ++v) {
      const int m = mBase + v + 8 * hi;
      out[m * D_MODEL + eBase + t * 16 + n16] = acc[t][v];
    }
}

// ---------- launch ----------

extern "C" void kernel_launch(void* const* d_in, const int* in_sizes, int n_in,
                              void* d_out, int out_size, void* d_ws, size_t ws_size,
                              hipStream_t stream) {
  const float* x  = (const float*)d_in[0];
  const float* wq = (const float*)d_in[1];
  const float* wk = (const float*)d_in[2];
  const float* wv = (const float*)d_in[3];
  const float* wo = (const float*)d_in[4];

  unsigned short* p = (unsigned short*)d_ws;
  unsigned short* xb  = p; p += (size_t)BS * D_MODEL;
  unsigned short* wqb = p; p += (size_t)D_MODEL * D_MODEL;
  unsigned short* wkb = p; p += (size_t)D_MODEL * D_MODEL;
  unsigned short* wvb = p; p += (size_t)D_MODEL * D_MODEL;
  unsigned short* wob = p; p += (size_t)D_MODEL * D_MODEL;
  unsigned short* qbf = p; p += (size_t)BS * D_MODEL;
  unsigned short* kbf = p; p += (size_t)BS * D_MODEL;
  unsigned short* vtb = p; p += (size_t)BS * D_MODEL;
  unsigned short* abf = p; p += (size_t)BS * D_MODEL;
  // total: 48 MB of workspace (fits in 192 MB L2)

  const int NX = BS * D_MODEL;          // 4,194,304
  const int NW = D_MODEL * D_MODEL;     // 1,048,576
  cvt_kernel<<<NX / 1024, 256, 0, stream>>>(x,  xb,  NX);
  cvt_kernel<<<NW / 1024, 256, 0, stream>>>(wq, wqb, NW);
  cvt_kernel<<<NW / 1024, 256, 0, stream>>>(wk, wkb, NW);
  cvt_kernel<<<NW / 1024, 256, 0, stream>>>(wv, wvb, NW);
  cvt_kernel<<<NW / 1024, 256, 0, stream>>>(wo, wob, NW);

  qkv_proj_kernel<<<dim3(BS / 128, D_MODEL / 64, 3), 256, 0, stream>>>(
      xb, wqb, wkb, wvb, qbf, kbf, vtb);

  attn_kernel<<<dim3(SEQ / 128, NHEAD, BATCH), 256, 0, stream>>>(
      qbf, kbf, vtb, abf);

  out_proj_kernel<<<dim3(BS / 128, D_MODEL / 64), 256, 0, stream>>>(
      abf, wob, (float*)d_out);
}
